// SemiMarkovModule_12120397710085
// MI455X (gfx1250) — matrix operand order repaired and checked by reading, not compiled, not run on hardware
//
#include <hip/hip_runtime.h>
#include <hip/hip_bf16.h>
#include <math.h>

#define BIG_NEG -1000000000.0f

constexpr int B = 16, N = 2048, D = 256, C = 128, K = 20;

typedef __attribute__((ext_vector_type(16))) __bf16 v16bf;
typedef __attribute__((ext_vector_type(8)))  float  v8f;
typedef __attribute__((ext_vector_type(4)))  unsigned int tdm_v4u;
typedef __attribute__((ext_vector_type(8)))  int          tdm_v8i;
typedef __attribute__((ext_vector_type(4)))  int          tdm_v4i;

// ---------------- workspace layout (float elements) ----------------
constexpr size_t EMIS_OFF  = 0;
constexpr size_t EMIS_SZ   = (size_t)B * N * C;               // 4,194,304
constexpr size_t CUM_OFF   = EMIS_OFF + EMIS_SZ;
constexpr size_t CUM_SZ    = (size_t)B * (N + 1) * C;         // 4,196,352
constexpr size_t XQ_OFF    = CUM_OFF + CUM_SZ;                // B*N
constexpr size_t MQ_OFF    = XQ_OFF + (size_t)B * N;          // C
constexpr size_t MW_OFF    = MQ_OFF + C;                      // C*D (means * inv_var)
constexpr size_t TT_OFF    = MW_OFF + (size_t)C * D;          // C*C  Tt[j*C+i] = exp(trans_lp[i][j])
constexpr size_t INIT_OFF  = TT_OFF + (size_t)C * C;          // C
constexpr size_t CONST_OFF = INIT_OFF + C;                    // 1

// ---------------- prep: constants, init_lp, mq, mw, exp(trans_lp) ----------------
__global__ __launch_bounds__(256) void prep_kernel(const float* __restrict__ means,
                                                   const float* __restrict__ cov,
                                                   const float* __restrict__ trans,
                                                   const float* __restrict__ initlg,
                                                   float* __restrict__ ws) {
  int tid = threadIdx.x;
  __shared__ float ilse;
  if (tid == 0) {
    float sl = 0.0f;
    for (int dd = 0; dd < D; ++dd) sl += __logf(cov[dd]);
    ws[CONST_OFF] = -0.5f * ((float)D * 1.8378770664093453f + sl);  // log(2*pi)
    float m = -3.4e38f;
    for (int i = 0; i < C; ++i) m = fmaxf(m, initlg[i]);
    float s = 0.0f;
    for (int i = 0; i < C; ++i) s += __expf(initlg[i] - m);
    ilse = m + __logf(s);
  }
  __syncthreads();
  if (tid < C) {
    ws[INIT_OFF + tid] = initlg[tid] - ilse;
    float s = 0.0f;
    for (int dd = 0; dd < D; ++dd) {
      float mv = means[tid * D + dd];
      s += mv * mv / cov[dd];
    }
    ws[MQ_OFF + tid] = s;
    // column j of trans log-softmax (axis=0), stored as exp, transposed
    int j = tid;
    float m = -3.4e38f;
    for (int i = 0; i < C; ++i) {
      float v = (i == j) ? BIG_NEG : trans[i * C + j];
      m = fmaxf(m, v);
    }
    float ss = 0.0f;
    for (int i = 0; i < C; ++i) {
      float v = (i == j) ? BIG_NEG : trans[i * C + j];
      ss += __expf(v - m);
    }
    float lse = m + __logf(ss);
    for (int i = 0; i < C; ++i) {
      float v = (i == j) ? BIG_NEG : trans[i * C + j];
      ws[TT_OFF + (size_t)j * C + i] = __expf(v - lse);  // Tt[j][i] = exp(trans_lp[i][j])
    }
  }
  for (int idx = tid; idx < C * D; idx += 256) {
    int dd = idx & (D - 1);
    ws[MW_OFF + idx] = means[idx] / cov[dd];
  }
}

// ---------------- xq[b,n] = sum_d f^2 * inv_var ----------------
__global__ __launch_bounds__(256) void xq_kernel(const float* __restrict__ feats,
                                                 const float* __restrict__ cov,
                                                 float* __restrict__ ws) {
  int idx = blockIdx.x * 256 + threadIdx.x;  // < B*N
  const float* f = feats + (size_t)idx * D;
  float s = 0.0f;
  for (int dd = 0; dd < D; ++dd) {
    float v = f[dd];
    s += v * v / cov[dd];
  }
  ws[XQ_OFF + idx] = s;
}

// ---------------- emission GEMM via bf16 WMMA ----------------
// grid: (N/16, B), block: 256 = 8 waves; wave w owns c-tile [16w, 16w+16)
__global__ __launch_bounds__(256) void emis_kernel(const float* __restrict__ feats,
                                                   float* __restrict__ ws) {
  const float* mw   = ws + MW_OFF;
  const float* mq   = ws + MQ_OFF;
  const float* xq   = ws + XQ_OFF;
  float*       emis = ws + EMIS_OFF;
  const float  cst  = ws[CONST_OFF];

  int b    = blockIdx.y;
  int n0   = blockIdx.x * 16;
  int lane = threadIdx.x & 31;
  int wid  = threadIdx.x >> 5;
  int c0   = wid * 16;
  int hi   = lane >> 4;       // half-wave select
  int l15  = lane & 15;

  int arow = n0 + l15;        // A matrix row (M)
  int bcol = c0 + l15;        // B matrix column (N)
  const float* Ap = feats + ((size_t)b * N + arow) * D;
  const float* Bp = mw + (size_t)bcol * D;

  v8f acc = {};
  #pragma unroll
  for (int kk = 0; kk < D; kk += 32) {
    // A 16-bit 16x32 layout: lanes0-15: elems0-7 = K 0..7, elems8-15 = K 16..23;
    // lanes16-31 shifted by +8 within each 16-K half.  (b128 vector loads)
    float4 a0 = *(const float4*)(Ap + kk + hi * 8);
    float4 a1 = *(const float4*)(Ap + kk + hi * 8 + 4);
    float4 a2 = *(const float4*)(Ap + kk + 16 + hi * 8);
    float4 a3 = *(const float4*)(Ap + kk + 16 + hi * 8 + 4);
    // B 32x16 layout: same K order as A (elems0-7 = K 8*hi..+7, elems8-15 = K 16+8*hi..+7; col = lane&15)
    float4 b0 = *(const float4*)(Bp + kk + hi * 8);
    float4 b1 = *(const float4*)(Bp + kk + hi * 8 + 4);
    float4 b2 = *(const float4*)(Bp + kk + 16 + hi * 8);
    float4 b3 = *(const float4*)(Bp + kk + 16 + hi * 8 + 4);

    v16bf a, bb;
    a[0]=(__bf16)a0.x;  a[1]=(__bf16)a0.y;  a[2]=(__bf16)a0.z;  a[3]=(__bf16)a0.w;
    a[4]=(__bf16)a1.x;  a[5]=(__bf16)a1.y;  a[6]=(__bf16)a1.z;  a[7]=(__bf16)a1.w;
    a[8]=(__bf16)a2.x;  a[9]=(__bf16)a2.y;  a[10]=(__bf16)a2.z; a[11]=(__bf16)a2.w;
    a[12]=(__bf16)a3.x; a[13]=(__bf16)a3.y; a[14]=(__bf16)a3.z; a[15]=(__bf16)a3.w;
    bb[0]=(__bf16)b0.x;  bb[1]=(__bf16)b0.y;  bb[2]=(__bf16)b0.z;  bb[3]=(__bf16)b0.w;
    bb[4]=(__bf16)b1.x;  bb[5]=(__bf16)b1.y;  bb[6]=(__bf16)b1.z;  bb[7]=(__bf16)b1.w;
    bb[8]=(__bf16)b2.x;  bb[9]=(__bf16)b2.y;  bb[10]=(__bf16)b2.z; bb[11]=(__bf16)b2.w;
    bb[12]=(__bf16)b3.x; bb[13]=(__bf16)b3.y; bb[14]=(__bf16)b3.z; bb[15]=(__bf16)b3.w;

    acc = __builtin_amdgcn_wmma_f32_16x16x32_bf16(false, a, false, bb,
                                                  (short)0, acc, false, false);
  }

  // C/D f32 16x16: VGPR r -> row n0+r (lanes0-15) or n0+r+8 (lanes16-31), col = c0+(lane&15)
  #pragma unroll
  for (int r = 0; r < 8; ++r) {
    int row = n0 + r + hi * 8;
    float val = cst - 0.5f * (xq[(size_t)b * N + row] + mq[bcol]) + acc[r];
    emis[((size_t)b * N + row) * C + bcol] = val;
  }
}

// ---------------- cumsum over n: cum[b][0..N][c] ----------------
__global__ __launch_bounds__(128) void cumsum_kernel(float* __restrict__ ws) {
  int b = blockIdx.x, c = threadIdx.x;
  const float* emis = ws + EMIS_OFF + (size_t)b * N * C;
  float*       cum  = ws + CUM_OFF + (size_t)b * (N + 1) * C;
  float run = 0.0f;
  cum[c] = 0.0f;
  for (int n = 0; n < N; ++n) {
    run += emis[(size_t)n * C + c];
    cum[(size_t)(n + 1) * C + c] = run;
  }
}

// ---------------- TDM: DMA one 128-float cum row into LDS ----------------
// D# per CDNA5 ISA (6-arg builtin on this toolchain).
__device__ __forceinline__ void tdm_load_row(const float* gsrc, unsigned lds_off) {
  unsigned long long ga = (unsigned long long)(size_t)gsrc;
  tdm_v4u g0;
  g0[0] = 1u;                                                  // count=1, user desc
  g0[1] = lds_off;                                             // lds_addr (bytes)
  g0[2] = (unsigned)ga;                                        // global_addr[31:0]
  g0[3] = (unsigned)((ga >> 32) & 0x1FFFFFFu) | 0x80000000u;   // addr[56:32] | type=2
  tdm_v8i g1;
  g1[0] = 0x00020000;        // wg_mask=0, data_size=2 (4 bytes)
  g1[1] = (int)(128u << 16); // tensor_dim0 = 128 (bits 79:48, low half)
  g1[2] = (int)(1u << 16);   // tensor_dim1 = 1   (bits 111:80, low half)
  g1[3] = (int)(128u << 16); // tile_dim0 = 128   (bits 127:112)
  g1[4] = 0;                 // tile_dim1/2 = 0 (1-D tile)
  g1[5] = 128;               // tensor_dim0_stride = 128
  g1[6] = 0;
  g1[7] = 0;
  tdm_v4i z4 = {0, 0, 0, 0};
  tdm_v8i z8 = {0, 0, 0, 0, 0, 0, 0, 0};
  __builtin_amdgcn_tensor_load_to_lds(g0, g1, z4, z4, z8, 0);
}

// ---------------- semi-Markov forward scan: 1 block per batch ----------------
__global__ __launch_bounds__(128) void scan_kernel(const int* __restrict__ lengths,
                                                   const float* __restrict__ plr,
                                                   const float* __restrict__ ws,
                                                   float* __restrict__ out) {
  constexpr int RING = 32;
  __shared__ float ring[RING * C];     // beta sliding window (rows mod 32)
  __shared__ float cumring[RING * C];  // cum sliding window, filled by TDM
  __shared__ float lenrev[K * C];
  __shared__ float p_s[C];
  __shared__ float wred[4];

  int c = threadIdx.x, b = blockIdx.x;
  const float* cum    = ws + CUM_OFF + (size_t)b * (N + 1) * C;
  const float* Tt     = ws + TT_OFF;   // Tt[j*C + c] = exp(trans_lp[c][j]) (L2-resident)
  const float* initlp = ws + INIT_OFF;
  bool issuer = (threadIdx.x < 32);    // wave 0 drives the TDM

  // len_rev[k][c] = len_lp[K-1-k][c], len_lp[j] = (j+1)*lr - exp(lr) - lgamma(j+2)
  float lr  = plr[c];
  float elr = __expf(lr);
  #pragma unroll
  for (int k = 0; k < K; ++k)
    lenrev[k * C + c] = (float)(K - k) * lr - elr - lgammaf((float)(K - k + 1));

  #pragma unroll
  for (int i = 0; i < K; ++i) ring[(i & (RING - 1)) * C + c] = BIG_NEG;  // rows 0..K-1
  ring[(K & (RING - 1)) * C + c] = initlp[c];                            // row K = init

  int lenb = lengths[b];
  if (lenb < 1) lenb = 1;
  if (lenb > N) lenb = N;

  // Prologue: DMA cum rows 0,1,2; rows 0,1 must land before t=1.
  if (issuer) {
    tdm_load_row(cum + 0 * C, (unsigned)(size_t)(cumring + 0 * C));
    tdm_load_row(cum + 1 * C, (unsigned)(size_t)(cumring + 1 * C));
    tdm_load_row(cum + 2 * C, (unsigned)(size_t)(cumring + 2 * C));
    __builtin_amdgcn_s_wait_tensorcnt(1);
  }
  __syncthreads();

  for (int t = 1; t <= lenb; ++t) {
    float ct = cumring[(t & (RING - 1)) * C + c];

    float sp[K];
    float m = -3.4e38f;
    #pragma unroll
    for (int k = 0; k < K; ++k) {
      int row = t + k;
      float w  = ring[(row & (RING - 1)) * C + c];
      int idx  = row - K;  // in [t-K, t-1]: resident in cumring when >= 0
      float cw = (idx >= 0) ? cumring[(idx & (RING - 1)) * C + c] : 0.0f;
      float v  = w + lenrev[k * C + c] + (ct - cw);
      sp[k] = v;
      m = fmaxf(m, v);
    }
    float s = 0.0f;
    #pragma unroll
    for (int k = 0; k < K; ++k) s += __expf(sp[k] - m);
    float alpha = m + __logf(s);

    // block max of alpha (wave32 shuffles + 4-way LDS combine)
    float am = alpha;
    #pragma unroll
    for (int off = 16; off >= 1; off >>= 1) am = fmaxf(am, __shfl_xor(am, off, 32));
    if ((c & 31) == 0) wred[c >> 5] = am;
    __syncthreads();
    float bm = fmaxf(fmaxf(wred[0], wred[1]), fmaxf(wred[2], wred[3]));

    p_s[c] = __expf(alpha - bm);
    __syncthreads();

    // beta_t[c] = log( sum_j exp(trans_lp[c][j]) * p[j] ) + bm
    float s2 = 0.0f;
    const float* Tc = Tt + c;
    for (int j = 0; j < C; ++j) s2 = fmaf(Tc[(size_t)j * C], p_s[j], s2);
    float bnew = __logf(fmaxf(s2, 1e-37f)) + bm;
    ring[((t + K) & (RING - 1)) * C + c] = bnew;

    if (t == lenb && c == 0) {  // out[b] = logsumexp_c alpha_t  at t = lengths[b]
      float ssum = 0.0f;
      for (int j = 0; j < C; ++j) ssum += p_s[j];
      out[b] = bm + __logf(ssum);
    }

    // Overlap: DMA row t+2 while step t finishes; ensure row t+1 has landed,
    // then the end-of-step barrier publishes it to all waves.
    if (issuer) {
      int nrow = t + 2;
      if (nrow <= N)
        tdm_load_row(cum + (size_t)nrow * C,
                     (unsigned)(size_t)(cumring + (size_t)(nrow & (RING - 1)) * C));
      __builtin_amdgcn_s_wait_tensorcnt(1);
    }
    __syncthreads();
  }
}

extern "C" void kernel_launch(void* const* d_in, const int* in_sizes, int n_in,
                              void* d_out, int out_size, void* d_ws, size_t ws_size,
                              hipStream_t stream) {
  const float* feats  = (const float*)d_in[0];
  const int*   lens   = (const int*)d_in[1];
  const float* means  = (const float*)d_in[2];
  const float* cov    = (const float*)d_in[3];
  const float* trans  = (const float*)d_in[4];
  const float* initlg = (const float*)d_in[5];
  const float* plr    = (const float*)d_in[6];
  float* out = (float*)d_out;
  float* ws  = (float*)d_ws;

  prep_kernel<<<1, 256, 0, stream>>>(means, cov, trans, initlg, ws);
  xq_kernel<<<(B * N) / 256, 256, 0, stream>>>(feats, cov, ws);
  emis_kernel<<<dim3(N / 16, B), 256, 0, stream>>>(feats, ws);
  cumsum_kernel<<<B, C, 0, stream>>>(ws);
  scan_kernel<<<B, C, 0, stream>>>(lens, plr, ws, out);
}
